// XCPNetImpl_75634374082667
// MI455X (gfx1250) — matrix operand order repaired and checked by reading, not compiled, not run on hardware
//
#include <hip/hip_runtime.h>
#include <hip/hip_bf16.h>

typedef __attribute__((ext_vector_type(16))) _Float16 v16h;
typedef __attribute__((ext_vector_type(8)))  _Float16 v8h;
typedef __attribute__((ext_vector_type(8)))  float    v8f;

#define E_NUM 4
#define NDIM  448
#define HID   240
#define B_FR  128
#define NAT   512
#define MAXTILES 36
#define KC0   14          // 448 / 32
#define KC1   8           // 256 / 32 (240 zero-padded to 256)
#define NT    15          // 240 / 16

// workspace byte offsets
#define WS_PERM    0                         // 576 ints
#define WS_TILETY  2304                      // 36 ints
#define WS_NTILES  2448                      // 1 int
#define WS_W0P     4096                      // 4*14*15*512 halves = 860160 B
#define WS_W1P     (4096 + 860160)           // 4*8*15*512 halves  = 491520 B
#define WS_W2P     (4096 + 860160 + 491520)

// ---------------------------------------------------------------- zero output
__global__ void zero_out_k(float* out) {
    if (threadIdx.x < B_FR) out[threadIdx.x] = 0.0f;
}

// ------------------------------------------------- build type-sorted atom perm
__global__ void prep_perm_k(const int* __restrict__ tmap, int* perm,
                            int* tile_type, int* ntiles) {
    if (threadIdx.x != 0 || blockIdx.x != 0) return;
    int pos = 0, t = 0;
    for (int e = 0; e < E_NUM; ++e) {
        int start = pos;
        for (int n = 0; n < NAT; ++n)
            if (tmap[n] == e) perm[pos++] = n;
        while (pos & 15) perm[pos++] = -1;      // pad tile with invalid rows
        for (int q = start; q < pos; q += 16) tile_type[t++] = e;
    }
    *ntiles = t;
    for (; t < MAXTILES; ++t) tile_type[t] = -1;
}

// ------------------------------------- pack f32 weights -> per-lane B-fragments
// Packed layout per expert: [kc][nt][lane(0..31)][j(0..15)] halves, where
// lane = (K-half select <<4) | N, element j holds W[kc*32 + hi*16 + j][nt*16+N].
__global__ void pack_weights_k(const float* __restrict__ W0,
                               const float* __restrict__ W1,
                               const float* __restrict__ W2,
                               _Float16* W0p, _Float16* W1p, _Float16* W2p) {
    int t = blockIdx.x * blockDim.x + threadIdx.x;
    const int N0 = E_NUM * KC0 * NT * 512;
    const int N1 = E_NUM * KC1 * NT * 512;
    if (t < N0) {
        int j = t & 15, lane = (t >> 4) & 31, nt = (t >> 9) % NT;
        int kc = (t / (512 * NT)) % KC0, e = t / (512 * NT * KC0);
        int n = nt * 16 + (lane & 15);
        int k = kc * 32 + ((lane >> 4) << 4) + j;
        W0p[t] = (_Float16)W0[(e * NDIM + k) * HID + n];
    } else if (t < N0 + 2 * N1) {
        int u = t - N0;
        const float* Ws = (u < N1) ? W1 : W2;
        _Float16*    Wd = (u < N1) ? W1p : W2p;
        int v = (u < N1) ? u : u - N1;
        int j = v & 15, lane = (v >> 4) & 31, nt = (v >> 9) % NT;
        int kc = (v / (512 * NT)) % KC1, e = v / (512 * NT * KC1);
        int n = nt * 16 + (lane & 15);
        int k = kc * 32 + ((lane >> 4) << 4) + j;
        Wd[v] = (k < HID) ? (_Float16)Ws[(e * HID + k) * HID + n] : (_Float16)0.0f;
    }
}

// ------------------------------------------------------------ fused WMMA layer
// src: 16 x LD halves in LDS (A, row-major), wp: packed B for this expert,
// dst: 16 x 256 halves in LDS.  D = relu(A*W + bias) (+ residual src).
template<int KC, int LD, bool RES>
__device__ __forceinline__ void glayer(const _Float16* __restrict__ src,
                                       const _Float16* __restrict__ wp,
                                       const float* __restrict__ bias,
                                       _Float16* __restrict__ dst, int lane) {
    const int nlo  = lane & 15;       // N column / A row for this lane
    const int hi   = lane >> 4;       // K-half select
    const int koff = hi * 16;         // A order used here: lanes16-31 start at K+16
    const int mrow = hi * 8;          // C layout: lanes16-31 hold M=8..15
#pragma unroll 1
    for (int nt = 0; nt < NT; ++nt) {
        float bv = bias[nt * 16 + nlo];
        v8f acc;
#pragma unroll
        for (int r = 0; r < 8; ++r) acc[r] = bv;
#pragma unroll 2
        for (int kc = 0; kc < KC; ++kc) {
            const _Float16* ap = src + nlo * LD + kc * 32 + koff;
            v8h lo = *(const v8h*)(ap);           // K = kb .. kb+7
            v8h hh = *(const v8h*)(ap + 8);       // K = kb+8 .. kb+15
            v16h a;
#pragma unroll
            for (int i = 0; i < 8; ++i) { a[i] = lo[i]; a[i + 8] = hh[i]; }
            v16h bf = *(const v16h*)(wp + (size_t)((kc * NT + nt) * 512) + lane * 16);
            acc = __builtin_amdgcn_wmma_f32_16x16x32_f16(
                      false, a, false, bf, (short)0, acc, false, false);
        }
        int col = nt * 16 + nlo;
#pragma unroll
        for (int r = 0; r < 8; ++r) {
            float v = acc[r] > 0.0f ? acc[r] : 0.0f;
            if (RES) v += (float)src[(mrow + r) * LD + col];
            dst[(mrow + r) * 256 + col] = (_Float16)v;
        }
    }
}

// ------------------------------------------------------------------ main fused
__global__ __launch_bounds__(32)
void moe_mlp_k(const float* __restrict__ x,
               const float* __restrict__ b0, const float* __restrict__ b1,
               const float* __restrict__ b2,
               const float* __restrict__ W3, const float* __restrict__ b3,
               const int* __restrict__ perm, const int* __restrict__ tile_type,
               const int* __restrict__ ntiles,
               const _Float16* __restrict__ W0p, const _Float16* __restrict__ W1p,
               const _Float16* __restrict__ W2p,
               float* __restrict__ out) {
    __shared__ _Float16 hX[16 * NDIM];   // layer-0 input (f16)
    __shared__ _Float16 hA[16 * 256];
    __shared__ _Float16 hB[16 * 256];
    __shared__ int   aidx[16];
    __shared__ float red[32];

    const int tile = blockIdx.x;
    if (tile >= *ntiles) return;                 // uniform -> EXEC stays full
    const int e    = tile_type[tile];
    const int b    = blockIdx.y;
    const int lane = threadIdx.x;

    if (lane < 16) aidx[lane] = perm[tile * 16 + lane];
    // zero the K pad columns (240..255) of both ping-pong buffers
    for (int i = lane; i < 512; i += 32) {
        _Float16* buf = (i & 256) ? hB : hA;
        buf[((i >> 4) & 15) * 256 + 240 + (i & 15)] = (_Float16)0.0f;
    }
    __syncthreads();

    // gather 16 atom rows of x, convert f32 -> f16 into LDS (invalid rows = 0)
    for (int i = lane; i < 16 * (NDIM / 4); i += 32) {
        int r = i / (NDIM / 4), c4 = (i % (NDIM / 4)) * 4;
        int atom = aidx[r];
        float4 v = make_float4(0.f, 0.f, 0.f, 0.f);
        if (atom >= 0)
            v = *(const float4*)&x[((size_t)b * NAT + atom) * NDIM + c4];
        hX[r * NDIM + c4 + 0] = (_Float16)v.x;
        hX[r * NDIM + c4 + 1] = (_Float16)v.y;
        hX[r * NDIM + c4 + 2] = (_Float16)v.z;
        hX[r * NDIM + c4 + 3] = (_Float16)v.w;
    }
    __syncthreads();

    glayer<KC0, NDIM, false>(hX, W0p + (size_t)e * KC0 * NT * 512, b0 + e * HID, hA, lane);
    __syncthreads();
    glayer<KC1, 256,  true >(hA, W1p + (size_t)e * KC1 * NT * 512, b1 + e * HID, hB, lane);
    __syncthreads();
    glayer<KC1, 256,  true >(hB, W2p + (size_t)e * KC1 * NT * 512, b2 + e * HID, hA, lane);
    __syncthreads();

    // final 240 -> 1 projection + masked sum over the 16 atoms of this tile
    const float* w3 = W3 + e * HID;
    float s = 0.0f;
    for (int m = 0; m < 16; ++m) {
        if (aidx[m] < 0) continue;
        for (int k = lane; k < HID; k += 32)
            s += (float)hA[m * 256 + k] * w3[k];
    }
    red[lane] = s;
    __syncthreads();
    if (lane == 0) {
        float tot = 0.0f;
        for (int i = 0; i < 32; ++i) tot += red[i];
        int nvalid = 0;
        for (int m = 0; m < 16; ++m) nvalid += (aidx[m] >= 0) ? 1 : 0;
        tot += (float)nvalid * b3[e];
        atomicAdd(&out[b], tot * (1.0f / (float)NAT));   // mean over atoms
    }
}

// -------------------------------------------------------------------- launcher
extern "C" void kernel_launch(void* const* d_in, const int* in_sizes, int n_in,
                              void* d_out, int out_size, void* d_ws, size_t ws_size,
                              hipStream_t stream) {
    const float* x  = (const float*)d_in[0];
    const float* W0 = (const float*)d_in[1];
    const float* b0 = (const float*)d_in[2];
    const float* W1 = (const float*)d_in[3];
    const float* b1 = (const float*)d_in[4];
    const float* W2 = (const float*)d_in[5];
    const float* b2 = (const float*)d_in[6];
    const float* W3 = (const float*)d_in[7];
    const float* b3 = (const float*)d_in[8];
    const int* tmap = (const int*)d_in[9];

    char* ws = (char*)d_ws;
    int* perm      = (int*)(ws + WS_PERM);
    int* tile_type = (int*)(ws + WS_TILETY);
    int* ntiles    = (int*)(ws + WS_NTILES);
    _Float16* W0p  = (_Float16*)(ws + WS_W0P);
    _Float16* W1p  = (_Float16*)(ws + WS_W1P);
    _Float16* W2p  = (_Float16*)(ws + WS_W2P);
    float* out     = (float*)d_out;

    zero_out_k<<<1, 128, 0, stream>>>(out);
    prep_perm_k<<<1, 1, 0, stream>>>(tmap, perm, tile_type, ntiles);

    const int npack = E_NUM * KC0 * NT * 512 + 2 * E_NUM * KC1 * NT * 512;
    pack_weights_k<<<(npack + 255) / 256, 256, 0, stream>>>(W0, W1, W2, W0p, W1p, W2p);

    dim3 grid(MAXTILES, B_FR);
    moe_mlp_k<<<grid, 32, 0, stream>>>(x, b0, b1, b2, W3, b3,
                                       perm, tile_type, ntiles,
                                       W0p, W1p, W2p, out);
}
